// TransformerEncoder_43705587204470
// MI455X (gfx1250) — matrix operand order repaired and checked by reading, not compiled, not run on hardware
//
#include <hip/hip_runtime.h>
#include <hip/hip_bf16.h>

// ---------- types ----------
typedef __attribute__((ext_vector_type(16))) __bf16 v16bf;
typedef __attribute__((ext_vector_type(8)))  float  v8f;

union Frag16 { v16bf v; unsigned int u[8]; };

// ---------- helpers ----------
__device__ __forceinline__ unsigned short f2bf(float f) {
    unsigned int u = __float_as_uint(f);
    u += 0x7fffu + ((u >> 16) & 1u);          // round-to-nearest-even
    return (unsigned short)(u >> 16);
}
__device__ __forceinline__ float bf2f(unsigned short h) {
    return __uint_as_float(((unsigned int)h) << 16);
}
__device__ __forceinline__ float act_apply(float x, int mode) {
    if (mode == 1) {                           // elu(x)+1  (phi feature map)
        return x > 0.f ? x + 1.f : __expf(x);
    }
    if (mode == 2) {                           // tanh-approx gelu
        float x3 = x * x * x;
        float t  = tanhf(0.7978845608028654f * (x + 0.044715f * x3));
        return 0.5f * x * (1.f + t);
    }
    return x;
}

// LDS byte offset of a __shared__ pointer (AS3 offset == low 32 bits of flat addr)
template <typename T>
__device__ __forceinline__ unsigned lds_off(const T* p) {
    return (unsigned)(uintptr_t)p;
}

// ---------- model dims ----------
#define BB   8
#define SS   2048
#define FF   256
#define MROWS (BB*SS)        // 16384
#define HH   8
#define DH   32
#define KFIN (SS*128)        // 262144
#define LAT  512
#define KCH  2048            // K-chunk of final projection
#define NCHUNK (KFIN/KCH)    // 128

// =====================================================================
// 1) x + pos  ->  bf16
// =====================================================================
__global__ __launch_bounds__(256) void addpos_kernel(
    const float* __restrict__ x, const float* __restrict__ pos,
    unsigned short* __restrict__ out)
{
    int i = blockIdx.x * 256 + threadIdx.x;          // 0 .. 16384*256-1
    out[i] = f2bf(x[i] + pos[i & (SS * FF - 1)]);
}

// =====================================================================
// 2) Tiled bf16 WMMA GEMM:  Out(bf16) = act( A(bf16)[M,K] @ W(f32)[K,N] + bias )
//    block tile 128x128, 8 waves of 32, wave tile 32x64 (2x4 wmma accums)
//    A tile staged with GLOBAL_LOAD_ASYNC_TO_LDS_B128 (ASYNCcnt path)
// =====================================================================
__global__ __launch_bounds__(256) void gemm_bf16_wmma(
    const unsigned short* __restrict__ A, const float* __restrict__ W,
    const float* __restrict__ bias, unsigned short* __restrict__ Out,
    int M, int N, int K, int act)
{
    __shared__ unsigned short Asm[128 * 40];   // 128 rows x 32 k  (pad to 40)
    __shared__ unsigned short Bsm[32 * 136];   // 32 k x 128 n     (pad to 136)

    const int tid  = threadIdx.x;
    const int m0   = blockIdx.y * 128;
    const int n0   = blockIdx.x * 128;
    const int wave = tid >> 5;
    const int lane = tid & 31;
    const int rw   = (wave >> 1) * 32;         // wave row base in block tile
    const int cw   = (wave & 1) * 64;          // wave col base in block tile

    v8f acc[2][4];
#pragma unroll
    for (int i = 0; i < 2; ++i)
#pragma unroll
        for (int j = 0; j < 4; ++j) {
            v8f z = {0.f,0.f,0.f,0.f,0.f,0.f,0.f,0.f};
            acc[i][j] = z;
        }

    const int arow = tid >> 2, acb = (tid & 3) * 8;       // A tile load map
    const int brow = tid >> 3, bcb = (tid & 7) * 16;      // B tile load map
    const unsigned a_lds = lds_off(&Asm[arow * 40 + acb]);// per-lane LDS dest (16B aligned)

    for (int kt = 0; kt < K; kt += 32) {
        // ---- stage A tile: async global->LDS copy (16B per lane) ----
        {
            unsigned long long ga =
                (unsigned long long)(A + (size_t)(m0 + arow) * K + kt + acb);
            asm volatile("global_load_async_to_lds_b128 %0, %1, off"
                         :: "v"(a_lds), "v"(ga) : "memory");
        }
        // ---- stage B tile (f32 -> bf16, 16 floats per thread) ----
        {
            const float4* src = (const float4*)(W + (size_t)(kt + brow) * N + n0 + bcb);
#pragma unroll
            for (int j = 0; j < 4; ++j) {
                float4 f = src[j];
                unsigned short* d = &Bsm[brow * 136 + bcb + j * 4];
                d[0] = f2bf(f.x); d[1] = f2bf(f.y); d[2] = f2bf(f.z); d[3] = f2bf(f.w);
            }
            // prefetch next K-step of W (global_prefetch_b8)
            if (kt + 32 < K)
                __builtin_prefetch(W + (size_t)(kt + 32 + brow) * N + n0 + bcb, 0, 0);
        }
        asm volatile("s_wait_asynccnt 0x0" ::: "memory");   // A-tile landed in LDS
        __syncthreads();

        // ---- build fragments per documented wave32 layouts ----
        Frag16 af[2], bfg[4];
        const int mrow  = lane & 15;
        const int khalf = (lane < 16) ? 0 : 8;
#pragma unroll
        for (int mt = 0; mt < 2; ++mt) {
            const unsigned short* arp = &Asm[(rw + mt * 16 + mrow) * 40];
#pragma unroll
            for (int v = 0; v < 8; ++v) {
                int k0 = (v < 4) ? (2 * v + khalf) : (16 + 2 * (v - 4) + khalf);
                af[mt].u[v] = *(const unsigned int*)&arp[k0];
            }
        }
#pragma unroll
        for (int nt = 0; nt < 4; ++nt) {
            const unsigned short* brp = &Bsm[cw + nt * 16 + mrow];
#pragma unroll
            for (int v = 0; v < 8; ++v) {
                int k0 = (v < 4) ? (2 * v + khalf) : (16 + 2 * (v - 4) + khalf);
                bfg[nt].u[v] = (unsigned int)brp[k0 * 136] | ((unsigned int)brp[(k0 + 1) * 136] << 16);
            }
        }

        // ---- 8 WMMAs ----
#pragma unroll
        for (int mt = 0; mt < 2; ++mt)
#pragma unroll
            for (int nt = 0; nt < 4; ++nt)
                acc[mt][nt] = __builtin_amdgcn_wmma_f32_16x16x32_bf16(
                    false, af[mt].v, false, bfg[nt].v,
                    (short)0, acc[mt][nt], false, false);
        __syncthreads();
    }

    // ---- epilogue: bias + activation + bf16 store ----
#pragma unroll
    for (int mt = 0; mt < 2; ++mt) {
#pragma unroll
        for (int nt = 0; nt < 4; ++nt) {
            const int n  = n0 + cw + nt * 16 + (lane & 15);
            const float bv = bias[n];
            const int mbase = m0 + rw + mt * 16 + ((lane < 16) ? 0 : 8);
#pragma unroll
            for (int v = 0; v < 8; ++v) {
                float val = act_apply(acc[mt][nt][v] + bv, act);
                Out[(size_t)(mbase + v) * N + n] = f2bf(val);
            }
        }
    }
}

// =====================================================================
// 3) kv[b,h,d,e] = sum_s phi(k)[b,s,h,d] * v[b,s,h,e] ;  z[b,h,d] = sum_s phi(k)
// =====================================================================
__global__ __launch_bounds__(256) void kvz_kernel(
    const unsigned short* __restrict__ Kp, const unsigned short* __restrict__ Vp,
    float* __restrict__ kv, float* __restrict__ z)
{
    const int bh = blockIdx.x;              // 0..63
    const int b = bh >> 3, h = bh & 7;
    __shared__ float ks[64][32];
    __shared__ float vs[64][32];
    const int tid = threadIdx.x;
    const int d = tid >> 3, eb = (tid & 7) * 4;
    float a0 = 0.f, a1 = 0.f, a2 = 0.f, a3 = 0.f, zacc = 0.f;

    for (int st = 0; st < SS; st += 64) {
#pragma unroll
        for (int j = 0; j < 8; ++j) {
            int i  = tid * 8 + j;           // 0..2047
            int sl = i >> 5, dd = i & 31;
            int g  = ((b * SS + st + sl) * FF) + h * DH + dd;
            ks[sl][dd] = bf2f(Kp[g]);
            vs[sl][dd] = bf2f(Vp[g]);
        }
        __syncthreads();
        for (int s = 0; s < 64; ++s) {
            float kd = ks[s][d];
            a0 += kd * vs[s][eb + 0];
            a1 += kd * vs[s][eb + 1];
            a2 += kd * vs[s][eb + 2];
            a3 += kd * vs[s][eb + 3];
            zacc += kd;
        }
        __syncthreads();
    }
    float* o = kv + bh * (DH * DH) + d * DH + eb;
    o[0] = a0; o[1] = a1; o[2] = a2; o[3] = a3;
    if ((tid & 7) == 0) z[bh * DH + d] = zacc;
}

// =====================================================================
// 4) attn = (q @ kv) / (q . z + 1e-6)
// =====================================================================
__global__ __launch_bounds__(256) void numden_kernel(
    const unsigned short* __restrict__ Q, const float* __restrict__ kv,
    const float* __restrict__ z, unsigned short* __restrict__ Attn)
{
    const int r = blockIdx.x;               // 0..16383
    const int b = r >> 11;
    __shared__ float qs[256];
    const int t = threadIdx.x;
    qs[t] = bf2f(Q[(size_t)r * FF + t]);
    __syncthreads();
    const int h = t >> 5, e = t & 31;
    const float* kvp = kv + ((b * HH + h) * DH * DH) + e;   // stride 32 along d
    const float* zp  = z + (b * HH + h) * DH;
    const float* qh  = qs + h * DH;
    float num = 0.f, den = 0.f;
#pragma unroll
    for (int d = 0; d < DH; ++d) {
        float qd = qh[d];
        num += qd * kvp[d * DH];
        den += qd * zp[d];
    }
    Attn[(size_t)r * FF + t] = f2bf(num / (den + 1e-6f));
}

// =====================================================================
// 5) streaming final projection, K-split into chunks (no atomics)
//    partial[mat][chunk][b][n] = sum_{k in chunk} xf[b,k] * W[k,n]
//    activations staged as f32 packed per-k: one 4B weight load + two
//    broadcast ds_load_b128 + 8 FMAs per k  ->  HBM-roofline bound
// =====================================================================
__global__ __launch_bounds__(256) void proj_kernel(
    const unsigned short* __restrict__ Xf,
    const float* __restrict__ Wmean, const float* __restrict__ Wdev,
    float* __restrict__ partial)
{
    const int mat = blockIdx.z;
    const float* W = mat ? Wdev : Wmean;
    const int n  = blockIdx.x * 256 + threadIdx.x;          // 0..511
    const int k0 = blockIdx.y * KCH;
    __shared__ float xsf[KCH * BB];                         // [kk][b], 64 KB
#pragma unroll
    for (int j = 0; j < (BB * KCH) / 256; ++j) {
        int i  = threadIdx.x + j * 256;
        int bb = i >> 11, kk = i & (KCH - 1);
        xsf[kk * BB + bb] = bf2f(Xf[(size_t)bb * KFIN + k0 + kk]);
    }
    __syncthreads();
    float acc[BB];
#pragma unroll
    for (int b = 0; b < BB; ++b) acc[b] = 0.f;
    const float4* xp = (const float4*)xsf;
#pragma unroll 2
    for (int kk = 0; kk < KCH; ++kk) {
        float w = W[(size_t)(k0 + kk) * LAT + n];           // coalesced along n
        float4 x0 = xp[kk * 2 + 0];                         // LDS broadcast
        float4 x1 = xp[kk * 2 + 1];
        acc[0] += x0.x * w; acc[1] += x0.y * w;
        acc[2] += x0.z * w; acc[3] += x0.w * w;
        acc[4] += x1.x * w; acc[5] += x1.y * w;
        acc[6] += x1.z * w; acc[7] += x1.w * w;
    }
    const size_t base = ((size_t)(mat * NCHUNK + blockIdx.y) * BB) * LAT + n;
#pragma unroll
    for (int b = 0; b < BB; ++b)
        partial[base + (size_t)b * LAT] = acc[b];
}

// =====================================================================
// 6) finalize: sum chunks + bias (exp for deviations)
// =====================================================================
__global__ __launch_bounds__(256) void finalize_kernel(
    const float* __restrict__ partial,
    const float* __restrict__ bmean, const float* __restrict__ bdev,
    float* __restrict__ out)
{
    const int g = blockIdx.x * 256 + threadIdx.x;           // 0..8191
    const int mat = g >> 12;
    const int loc = g & 4095;
    const int b = loc >> 9, n = loc & 511;
    float s = 0.f;
    for (int c = 0; c < NCHUNK; ++c)
        s += partial[((size_t)(mat * NCHUNK + c) * BB + b) * LAT + n];
    s += (mat ? bdev : bmean)[n];
    out[g] = mat ? __expf(s) : s;
}

// =====================================================================
// host-side launch
// =====================================================================
extern "C" void kernel_launch(void* const* d_in, const int* in_sizes, int n_in,
                              void* d_out, int out_size, void* d_ws, size_t ws_size,
                              hipStream_t stream) {
    (void)in_sizes; (void)n_in; (void)out_size; (void)ws_size;
    const float* x   = (const float*)d_in[0];
    const float* pos = (const float*)d_in[1];

    char* ws = (char*)d_ws;
    const size_t MB8 = (size_t)MROWS * FF * sizeof(unsigned short);   // 8 MB
    unsigned short* buf0 = (unsigned short*)(ws + 0 * MB8);
    unsigned short* bufQ = (unsigned short*)(ws + 1 * MB8);
    unsigned short* bufK = (unsigned short*)(ws + 2 * MB8);
    unsigned short* bufV = (unsigned short*)(ws + 3 * MB8);
    unsigned short* bufA = (unsigned short*)(ws + 4 * MB8);
    unsigned short* bufT = (unsigned short*)(ws + 5 * MB8);
    float* kvb     = (float*)(ws + 6 * MB8);
    float* zb      = (float*)(ws + 6 * MB8 + (size_t)(BB * HH * DH * DH) * 4);
    float* partial = (float*)(ws + 6 * MB8 + (size_t)(BB * HH * DH * DH) * 4
                                          + (size_t)(BB * HH * DH) * 4);

    addpos_kernel<<<MROWS * FF / 256, 256, 0, stream>>>(x, pos, buf0);

    for (int blk = 0; blk < 2; ++blk) {
        const int wb = 2 + blk * 12;
        const float* qw  = (const float*)d_in[wb + 0];
        const float* qb  = (const float*)d_in[wb + 1];
        const float* kw  = (const float*)d_in[wb + 2];
        const float* kb  = (const float*)d_in[wb + 3];
        const float* vw  = (const float*)d_in[wb + 4];
        const float* vb  = (const float*)d_in[wb + 5];
        const float* ow  = (const float*)d_in[wb + 6];
        const float* ob  = (const float*)d_in[wb + 7];
        const float* d1w = (const float*)d_in[wb + 8];
        const float* d1b = (const float*)d_in[wb + 9];
        const float* d2w = (const float*)d_in[wb + 10];
        const float* d2b = (const float*)d_in[wb + 11];
        const int Nd2 = (blk == 0) ? 256 : 128;

        dim3 g256(256 / 128, MROWS / 128);
        gemm_bf16_wmma<<<g256, 256, 0, stream>>>(buf0, qw, qb, bufQ, MROWS, 256, 256, 1);
        gemm_bf16_wmma<<<g256, 256, 0, stream>>>(buf0, kw, kb, bufK, MROWS, 256, 256, 1);
        gemm_bf16_wmma<<<g256, 256, 0, stream>>>(buf0, vw, vb, bufV, MROWS, 256, 256, 0);

        kvz_kernel<<<BB * HH, 256, 0, stream>>>(bufK, bufV, kvb, zb);
        numden_kernel<<<MROWS, 256, 0, stream>>>(bufQ, kvb, zb, bufA);

        gemm_bf16_wmma<<<g256, 256, 0, stream>>>(bufA, ow, ob, bufT, MROWS, 256, 256, 0);
        gemm_bf16_wmma<<<g256, 256, 0, stream>>>(bufT, d1w, d1b, bufA, MROWS, 256, 256, 2);
        dim3 gd2(Nd2 / 128, MROWS / 128);
        gemm_bf16_wmma<<<gd2, 256, 0, stream>>>(bufA, d2w, d2b, buf0, MROWS, Nd2, 256, 2);
    }

    const float* meanw = (const float*)d_in[26];
    const float* meanb = (const float*)d_in[27];
    const float* devw  = (const float*)d_in[28];
    const float* devb  = (const float*)d_in[29];

    proj_kernel<<<dim3(LAT / 256, NCHUNK, 2), 256, 0, stream>>>(buf0, meanw, devw, partial);
    finalize_kernel<<<2 * BB * LAT / 256, 256, 0, stream>>>(partial, meanb, devb, (float*)d_out);
}
